// _DeepseekV2ExpertMLPFP8_19095424598383
// MI455X (gfx1250) — matrix operand order repaired and checked by reading, not compiled, not run on hardware
//
#include <hip/hip_runtime.h>
#include <hip/hip_bf16.h>

typedef __attribute__((ext_vector_type(16))) _Float16 v16h;
typedef __attribute__((ext_vector_type(8)))  _Float16 v8h;
typedef __attribute__((ext_vector_type(8)))  float    v8f;
typedef __attribute__((ext_vector_type(4)))  float    v4f;

#define T_DIM 8192
#define H_DIM 5120
#define F_DIM 1536

__device__ __forceinline__ v16h cat8(v8h lo, v8h hi) {
  return __builtin_shufflevector(lo, hi, 0,1,2,3,4,5,6,7,8,9,10,11,12,13,14,15);
}

__device__ __forceinline__ v8f wmma_f16(v16h a, v16h b, v8f c) {
  // (neg_a, A, neg_b, B, c_mod, C, reuse_a, reuse_b)
  return __builtin_amdgcn_wmma_f32_16x16x32_f16(false, a, false, b, (short)0, c,
                                                false, false);
}

// CDNA5 async global->LDS copy (ASYNCcnt-tracked, no VGPR round trip).
// vDst = per-lane LDS byte offset, vAddr = per-lane 64-bit global address.
__device__ __forceinline__ void async_b128(unsigned lds_byte, const _Float16* g) {
  asm volatile("global_load_async_to_lds_b128 %0, %1, off"
               :: "v"(lds_byte), "v"((unsigned long long)(size_t)g)
               : "memory");
}
__device__ __forceinline__ void wait_async0() {
  asm volatile("s_wait_asynccnt 0x0" ::: "memory");
}

// ---------------- f32 -> f16 convert (x) ----------------
__global__ __launch_bounds__(256)
void cvt_f16_kernel(const float* __restrict__ src, _Float16* __restrict__ dst,
                    long n4) {
  long i = ((long)blockIdx.x * 256 + threadIdx.x);
  if (i >= n4) return;
  long idx = i * 4;
  v4f v = *(const v4f*)(src + idx);
  dst[idx + 0] = (_Float16)v[0];
  dst[idx + 1] = (_Float16)v[1];
  dst[idx + 2] = (_Float16)v[2];
  dst[idx + 3] = (_Float16)v[3];
}

// ------------- block-dequant + f32 -> f16 (weights) -------------
// w: [O, I] f32, s: [O/128, I/128]; out f16 = w * s[o/128][i/128]
__global__ __launch_bounds__(256)
void dequant_kernel(const float* __restrict__ w, const float* __restrict__ s,
                    _Float16* __restrict__ wh, int O, int I) {
  long i = ((long)blockIdx.x * 256 + threadIdx.x) * 4;
  if (i >= (long)O * I) return;
  int col = (int)(i % I);
  int row = (int)(i / I);
  float sc = s[(row >> 7) * (I >> 7) + (col >> 7)];  // 128%4==0: no block crossing
  v4f v = *(const v4f*)(w + i);
  wh[i + 0] = (_Float16)(v[0] * sc);
  wh[i + 1] = (_Float16)(v[1] * sc);
  wh[i + 2] = (_Float16)(v[2] * sc);
  wh[i + 3] = (_Float16)(v[3] * sc);
}

// ------------- GEMM1: h = silu(x Wg^T) * (x Wu^T), f16 out -------------
// tile: M=128 (tokens) x N=64 (features), BK=64; 8 waves in 4(M)x2(N) grid,
// each wave 32x32 = 2x2 WMMA fragments per matrix (gate + up).
// Double-buffered LDS fed by global_load_async_to_lds_b128.
__global__ __launch_bounds__(256)
void gateup_kernel(const _Float16* __restrict__ xh,
                   const _Float16* __restrict__ wgh,
                   const _Float16* __restrict__ wuh,
                   _Float16* __restrict__ hbuf) {
  extern __shared__ _Float16 smem[];
  // per-stage layout (halfs): xs[128][72] @0, gs[64][72] @9216, us[64][72] @13824
  constexpr int STAGE  = 18432;   // halfs per stage (36864 B)
  constexpr int GS_OFF = 9216;
  constexpr int US_OFF = 13824;

  const int tid  = threadIdx.x;
  const int lane = tid & 31;
  const int wave = tid >> 5;
  const int wm   = (wave & 3) * 32;
  const int wn   = (wave >> 2) * 32;
  const int t0   = blockIdx.x * 128;
  const int f0   = blockIdx.y * 64;
  const int l16  = lane & 15;
  const int hi   = lane >> 4;

  v8f accG[2][2], accU[2][2];
#pragma unroll
  for (int a = 0; a < 2; ++a)
#pragma unroll
    for (int b = 0; b < 2; ++b) { accG[a][b] = (v8f)0.0f; accU[a][b] = (v8f)0.0f; }

  const int xr = tid >> 3, xc = (tid & 7) * 8;  // staging coordinates

  auto issue_stage = [&](int kt, int buf) {
    const int k0 = kt * 64;
    const unsigned sb = (unsigned)(buf * STAGE) * 2u;  // stage base, bytes
#pragma unroll
    for (int i = 0; i < 4; ++i) {            // x tile: 128 rows x 64 halfs
      int r = xr + i * 32;
      async_b128(sb + (unsigned)(r * 72 + xc) * 2u,
                 xh + (long)(t0 + r) * H_DIM + k0 + xc);
    }
#pragma unroll
    for (int i = 0; i < 2; ++i) {            // gate/up tiles: 64 rows x 64 halfs
      int r = xr + i * 32;
      async_b128(sb + (unsigned)(GS_OFF + r * 72 + xc) * 2u,
                 wgh + (long)(f0 + r) * H_DIM + k0 + xc);
      async_b128(sb + (unsigned)(US_OFF + r * 72 + xc) * 2u,
                 wuh + (long)(f0 + r) * H_DIM + k0 + xc);
    }
  };

  issue_stage(0, 0);

  for (int kt = 0; kt < H_DIM / 64; ++kt) {
    const int buf = kt & 1;
    wait_async0();        // own async loads for stage kt landed in LDS
    __syncthreads();      // all waves' loads visible
    if (kt + 1 < H_DIM / 64)
      issue_stage(kt + 1, buf ^ 1);  // other buffer's readers done (prev barrier)

    const _Float16* xsp = smem + buf * STAGE;
    const _Float16* gsp = xsp + GS_OFF;
    const _Float16* usp = xsp + US_OFF;

#pragma unroll
    for (int kk = 0; kk < 64; kk += 32) {
      // A fragments: per-lane K halfs {g*8..g*8+7, 16+g*8..16+g*8+7}
      v16h afr[2];
#pragma unroll
      for (int im = 0; im < 2; ++im) {
        const _Float16* p = xsp + (wm + im * 16 + l16) * 72 + kk + hi * 8;
        afr[im] = cat8(*(const v8h*)p, *(const v8h*)(p + 16));
      }
#pragma unroll
      for (int in = 0; in < 2; ++in) {
        // B fragments: per-lane K halfs {g*8..g*8+7, 16+g*8..16+g*8+7}
        const _Float16* pg = gsp + (wn + in * 16 + l16) * 72 + kk + hi * 8;
        v16h bg = cat8(*(const v8h*)pg, *(const v8h*)(pg + 16));
        const _Float16* pu = usp + (wn + in * 16 + l16) * 72 + kk + hi * 8;
        v16h bu = cat8(*(const v8h*)pu, *(const v8h*)(pu + 16));
#pragma unroll
        for (int im = 0; im < 2; ++im) {
          accG[im][in] = wmma_f16(afr[im], bg, accG[im][in]);
          accU[im][in] = wmma_f16(afr[im], bu, accU[im][in]);
        }
      }
    }
    __syncthreads();      // reads of buf done before it is refilled
  }

  // epilogue: h = silu(gate) * up, store f16
#pragma unroll
  for (int im = 0; im < 2; ++im)
#pragma unroll
    for (int in = 0; in < 2; ++in) {
      const int n = f0 + wn + in * 16 + l16;
#pragma unroll
      for (int r = 0; r < 8; ++r) {
        const int m = t0 + wm + im * 16 + hi * 8 + r;
        float g = accG[im][in][r];
        float u = accU[im][in][r];
        float sv = g / (1.0f + __expf(-g));
        hbuf[(long)m * F_DIM + n] = (_Float16)(sv * u);
      }
    }
}

// ------------- GEMM2: out = h Wd^T, f32 out -------------
__global__ __launch_bounds__(256)
void down_kernel(const _Float16* __restrict__ hbuf,
                 const _Float16* __restrict__ wdh,
                 float* __restrict__ out) {
  extern __shared__ _Float16 smem[];
  // per-stage layout (halfs): hs[128][72] @0, ds[64][72] @9216
  constexpr int STAGE  = 13824;   // halfs per stage (27648 B)
  constexpr int DS_OFF = 9216;

  const int tid  = threadIdx.x;
  const int lane = tid & 31;
  const int wave = tid >> 5;
  const int wm   = (wave & 3) * 32;
  const int wn   = (wave >> 2) * 32;
  const int t0   = blockIdx.x * 128;
  const int n0   = blockIdx.y * 64;
  const int l16  = lane & 15;
  const int hi   = lane >> 4;

  v8f acc[2][2];
#pragma unroll
  for (int a = 0; a < 2; ++a)
#pragma unroll
    for (int b = 0; b < 2; ++b) acc[a][b] = (v8f)0.0f;

  const int xr = tid >> 3, xc = (tid & 7) * 8;

  auto issue_stage = [&](int kt, int buf) {
    const int k0 = kt * 64;
    const unsigned sb = (unsigned)(buf * STAGE) * 2u;
#pragma unroll
    for (int i = 0; i < 4; ++i) {
      int r = xr + i * 32;
      async_b128(sb + (unsigned)(r * 72 + xc) * 2u,
                 hbuf + (long)(t0 + r) * F_DIM + k0 + xc);
    }
#pragma unroll
    for (int i = 0; i < 2; ++i) {
      int r = xr + i * 32;
      async_b128(sb + (unsigned)(DS_OFF + r * 72 + xc) * 2u,
                 wdh + (long)(n0 + r) * F_DIM + k0 + xc);
    }
  };

  issue_stage(0, 0);

  for (int kt = 0; kt < F_DIM / 64; ++kt) {
    const int buf = kt & 1;
    wait_async0();
    __syncthreads();
    if (kt + 1 < F_DIM / 64)
      issue_stage(kt + 1, buf ^ 1);

    const _Float16* hsp = smem + buf * STAGE;
    const _Float16* dsp = hsp + DS_OFF;

#pragma unroll
    for (int kk = 0; kk < 64; kk += 32) {
      v16h afr[2];
#pragma unroll
      for (int im = 0; im < 2; ++im) {
        const _Float16* p = hsp + (wm + im * 16 + l16) * 72 + kk + hi * 8;
        afr[im] = cat8(*(const v8h*)p, *(const v8h*)(p + 16));
      }
#pragma unroll
      for (int in = 0; in < 2; ++in) {
        const _Float16* pb = dsp + (wn + in * 16 + l16) * 72 + kk + hi * 8;
        v16h bfr = cat8(*(const v8h*)pb, *(const v8h*)(pb + 16));
#pragma unroll
        for (int im = 0; im < 2; ++im)
          acc[im][in] = wmma_f16(afr[im], bfr, acc[im][in]);
      }
    }
    __syncthreads();
  }

#pragma unroll
  for (int im = 0; im < 2; ++im)
#pragma unroll
    for (int in = 0; in < 2; ++in) {
      const int n = n0 + wn + in * 16 + l16;
#pragma unroll
      for (int r = 0; r < 8; ++r) {
        const int m = t0 + wm + im * 16 + hi * 8 + r;
        out[(long)m * H_DIM + n] = acc[im][in][r];
      }
    }
}

extern "C" void kernel_launch(void* const* d_in, const int* in_sizes, int n_in,
                              void* d_out, int out_size, void* d_ws, size_t ws_size,
                              hipStream_t stream) {
  (void)in_sizes; (void)n_in; (void)out_size; (void)ws_size;
  const float* x      = (const float*)d_in[0];
  const float* w_gate = (const float*)d_in[1];
  const float* s_gate = (const float*)d_in[2];
  const float* w_up   = (const float*)d_in[3];
  const float* s_up   = (const float*)d_in[4];
  const float* w_down = (const float*)d_in[5];
  const float* s_down = (const float*)d_in[6];
  float* out = (float*)d_out;

  // workspace layout (bytes)
  char* ws = (char*)d_ws;
  const size_t XH_BYTES = (size_t)T_DIM * H_DIM * 2;  // 83,886,080
  const size_t W_BYTES  = (size_t)F_DIM * H_DIM * 2;  // 15,728,640
  _Float16* xh   = (_Float16*)(ws);
  _Float16* wgh  = (_Float16*)(ws + XH_BYTES);
  _Float16* wuh  = (_Float16*)(ws + XH_BYTES + W_BYTES);
  _Float16* wdh  = (_Float16*)(ws + XH_BYTES + 2 * W_BYTES);
  _Float16* hbuf = (_Float16*)(ws + XH_BYTES + 3 * W_BYTES);

  // 1) convert x to f16
  {
    long n4 = (long)T_DIM * H_DIM / 4;
    cvt_f16_kernel<<<(unsigned)((n4 + 255) / 256), 256, 0, stream>>>(x, xh, n4);
  }
  // 2) dequant weights into f16
  {
    long n4 = (long)F_DIM * H_DIM / 4;
    unsigned nb = (unsigned)((n4 + 255) / 256);
    dequant_kernel<<<nb, 256, 0, stream>>>(w_gate, s_gate, wgh, F_DIM, H_DIM);
    dequant_kernel<<<nb, 256, 0, stream>>>(w_up,   s_up,   wuh, F_DIM, H_DIM);
    dequant_kernel<<<nb, 256, 0, stream>>>(w_down, s_down, wdh, H_DIM, F_DIM);
  }
  // 3) fused gate/up GEMM + silu*mul -> h (f16); double-buffered async LDS
  {
    dim3 grid(T_DIM / 128, F_DIM / 64);
    gateup_kernel<<<grid, 256, 2 * 36864, stream>>>(xh, wgh, wuh, hbuf);
  }
  // 4) down GEMM -> out (f32); double-buffered async LDS
  {
    dim3 grid(T_DIM / 128, H_DIM / 64);
    down_kernel<<<grid, 256, 2 * 27648, stream>>>(hbuf, wdh, out);
  }
}
